// Graph_attention_6571299963150
// MI455X (gfx1250) — compile-verified, matrix operand order checked by reading, not run on hardware
//
#include <hip/hip_runtime.h>
#include <hip/hip_bf16.h>

typedef __attribute__((ext_vector_type(16))) _Float16 v16h;
typedef __attribute__((ext_vector_type(8)))  _Float16 v8h;
typedef __attribute__((ext_vector_type(8)))  float    v8f;
typedef int v4i __attribute__((vector_size(16)));

#define BS 4
#define NSEQ 512
#define DMODEL 1024
#define NH 16
#define HD 64

#if __has_builtin(__builtin_amdgcn_global_load_async_to_lds_b128)
#define USE_ASYNC_LDS 1
// builtin expects (v4i addrspace(1)*, v4i addrspace(3)*, imm offset, imm cpol)
#define ASG(p) ((__attribute__((address_space(1))) v4i*)(p))
#define ASL(p) ((__attribute__((address_space(3))) v4i*)(p))
#else
#define USE_ASYNC_LDS 0
#endif

// ---------------------------------------------------------------------------
// Fragment loader for 16-bit A/B of v_wmma_*_16x16x32_f16 (wave32).
// Per ISA 16-bit A 16x32 table: lane l holds row (l&15); half-index t maps to
//   K = g*8 + t (t=0..7),  K = g*8 + 16 + (t-8) (t=8..15),  g = l>>4.
// B (32x16) mirrors this with lane = column (per the sparse-B layout tables).
// ---------------------------------------------------------------------------
__device__ __forceinline__ v16h load_frag(const _Float16* rowptr, int kc, int g) {
    const _Float16* p0 = rowptr + kc * 32 + g * 8;
    union { v16h v; v8h h[2]; } u;
    u.h[0] = *(const v8h*)(p0);        // 16B load
    u.h[1] = *(const v8h*)(p0 + 16);   // 16B load
    return u.v;
}

// --------------------------- conversion kernels ----------------------------
__global__ void cvt_x_kernel(const float* __restrict__ x, _Float16* __restrict__ xh, int n) {
    int i = blockIdx.x * 256 + threadIdx.x;
    if (i < n) xh[i] = (_Float16)x[i];
}

// wT[c][e] = (f16) w[e][c]   (D x D)
__global__ void cvt_wT_kernel(const float* __restrict__ w, _Float16* __restrict__ wT) {
    int i = blockIdx.x * 256 + threadIdx.x;   // i = c*1024 + e
    int c = i >> 10, e = i & 1023;
    wT[i] = (_Float16)w[e * DMODEL + c];
}

// Fold wv (DxD) and wo (Dx1) into an f16 [NH][DMODEL] B-matrix:
// wvph[h][e] = sum_d wv[e][h*64+d]*wo[h*64+d];  bvp[h] = sum_d bv[h*64+d]*wo[h*64+d]
__global__ void fold_wv_kernel(const float* __restrict__ wv, const float* __restrict__ wo,
                               const float* __restrict__ bv, _Float16* __restrict__ wvph,
                               float* __restrict__ bvp) {
    int i = blockIdx.x * 256 + threadIdx.x;   // i = h*1024 + e  (16384 total)
    int h = i >> 10, e = i & 1023;
    float acc = 0.f;
    #pragma unroll 8
    for (int d = 0; d < HD; ++d) acc += wv[e * DMODEL + h * HD + d] * wo[h * HD + d];
    wvph[i] = (_Float16)acc;
    if (e == 0) {
        float bacc = 0.f;
        for (int d = 0; d < HD; ++d) bacc += bv[h * HD + d] * wo[h * HD + d];
        bvp[h] = bacc;
    }
}

// vp[b][h][j] via WMMA: (2048 x 1024) x (1024 x 16). One wave per 16-row tile.
__global__ __launch_bounds__(256) void vp_wmma_kernel(const _Float16* __restrict__ xh,
                                                      const _Float16* __restrict__ wvph,
                                                      const float* __restrict__ bvp,
                                                      float* __restrict__ vp) {
    int tid = threadIdx.x, lane = tid & 31, wid = tid >> 5;
    int tile = blockIdx.x * 8 + wid;          // 128 tiles
    int m0 = tile * 16;
    int lh = lane & 15, g = lane >> 4;
    const _Float16* arow = xh + ((size_t)(m0 + lh)) * DMODEL;
    const _Float16* brow = wvph + (size_t)lh * DMODEL;   // col h = lane
    v8f c = {};
    for (int kc = 0; kc < 32; ++kc) {
        v16h a = load_frag(arow, kc, g);
        v16h b = load_frag(brow, kc, g);
        c = __builtin_amdgcn_wmma_f32_16x16x32_f16(false, a, false, b, (short)0, c,
                                                   false, false);
    }
    #pragma unroll
    for (int v = 0; v < 8; ++v) {
        int row = m0 + v + 8 * g;
        int b = row >> 9, j = row & 511;
        vp[(((size_t)(b * NH + lh)) << 9) + j] = c[v] + bvp[lh];
    }
}

// ----------------------- WMMA projection GEMM (Q / K) ----------------------
// 2x2 register-blocked: each wave computes a 32x32 tile (4 WMMA chains),
// halving A/B fragment traffic per output element.
__global__ __launch_bounds__(256) void proj_kernel(const _Float16* __restrict__ A,
                                                   const _Float16* __restrict__ BT,
                                                   const float* __restrict__ bias,
                                                   _Float16* __restrict__ out) {
    int tid  = threadIdx.x;
    int lane = tid & 31;
    int wid  = tid >> 5;
    int tile = blockIdx.x * 8 + wid;          // 2048 wave-tiles (64 x 32)
    int mt = tile >> 5, nt = tile & 31;
    int m0 = mt * 32,  n0 = nt * 32;
    int lh = lane & 15, g = lane >> 4;

    const _Float16* arow0 = A  + ((size_t)(m0 + lh)) * DMODEL;
    const _Float16* arow1 = arow0 + (size_t)16 * DMODEL;
    const _Float16* brow0 = BT + ((size_t)(n0 + lh)) * DMODEL;
    const _Float16* brow1 = brow0 + (size_t)16 * DMODEL;

    v8f c00 = {}, c01 = {}, c10 = {}, c11 = {};
    for (int kc = 0; kc < 32; ++kc) {
        v16h a0 = load_frag(arow0, kc, g);
        v16h a1 = load_frag(arow1, kc, g);
        v16h b0 = load_frag(brow0, kc, g);
        v16h b1 = load_frag(brow1, kc, g);
        c00 = __builtin_amdgcn_wmma_f32_16x16x32_f16(false, a0, false, b0, (short)0, c00, false, false);
        c01 = __builtin_amdgcn_wmma_f32_16x16x32_f16(false, a0, false, b1, (short)0, c01, false, false);
        c10 = __builtin_amdgcn_wmma_f32_16x16x32_f16(false, a1, false, b0, (short)0, c10, false, false);
        c11 = __builtin_amdgcn_wmma_f32_16x16x32_f16(false, a1, false, b1, (short)0, c11, false, false);
    }
    v8f* cs[4] = { &c00, &c01, &c10, &c11 };
    #pragma unroll
    for (int sm = 0; sm < 2; ++sm) {
        #pragma unroll
        for (int sn = 0; sn < 2; ++sn) {
            v8f& c = *cs[sm * 2 + sn];
            #pragma unroll
            for (int v = 0; v < 8; ++v) {
                int row = m0 + sm * 16 + v + 8 * g;
                int col = n0 + sn * 16 + lh;
                float val = c[v] + bias[col];
                int bidx = row >> 9, i = row & 511;
                int h = col >> 6,   d = col & 63;
                out[(((size_t)(bidx * NH + h)) * NSEQ + i) * HD + d] = (_Float16)val;
            }
        }
    }
}

// ------------------------------ attention ----------------------------------
// One block per (b, 16-row i-tile). 8 waves, 256 threads.
// LDS: logits strip + s accumulator + async-staged bias strip (96 KB / 320 KB).
__global__ __launch_bounds__(256) void attn_kernel(const _Float16* __restrict__ Qh,
                                                   const _Float16* __restrict__ Kh,
                                                   const float* __restrict__ vp,
                                                   const float* __restrict__ bias,
                                                   const int*   __restrict__ mask,
                                                   const float* __restrict__ rin,
                                                   const float* __restrict__ bo,
                                                   float* __restrict__ out) {
    __shared__ float lds_logits[16 * NSEQ];
    __shared__ float lds_s[16 * NSEQ];
    __shared__ float lds_bias[16 * NSEQ];

    int tid  = threadIdx.x;
    int lane = tid & 31;
    int wid  = tid >> 5;
    int blk  = blockIdx.x;
    int b    = blk >> 5;
    int i0   = (blk & 31) * 16;
    int lh   = lane & 15, g = lane >> 4;

    // Per-thread fixed (row, col-set) across all heads -> load mask ONCE as bits.
    int rr = tid >> 4;                 // softmax row 0..15 (one 16-lane half-wave)
    int cb = tid & 15;
    int irow = i0 + rr;
    const int* maskrow = mask + ((size_t)b * NSEQ + irow) * NSEQ;
    unsigned mbits = 0;
    #pragma unroll
    for (int k = 0; k < 32; ++k) {
        int j = cb + (k << 4);
        if (__builtin_nontemporal_load(&maskrow[j]) != 0) mbits |= (1u << k);
    }

    for (int idx = tid; idx < 16 * NSEQ; idx += 256) lds_s[idx] = 0.f;
    __syncthreads();

    for (int h = 0; h < NH; ++h) {
        // ---- stage this head's contiguous 16x512 f32 bias strip into LDS,
        //      overlapped with the WMMA phase (async-to-LDS, ASYNCcnt) ----
        const float* bsrc = bias + (((size_t)(b * NH + h)) * NSEQ + i0) * NSEQ;
        #pragma unroll
        for (int it = 0; it < 8; ++it) {
            int e4 = (tid + it * 256) * 4;           // 16B per thread-iter
#if USE_ASYNC_LDS
            __builtin_amdgcn_global_load_async_to_lds_b128(ASG(bsrc + e4),
                                                           ASL(&lds_bias[e4]), 0, 0);
#else
            lds_bias[e4 + 0] = bsrc[e4 + 0];
            lds_bias[e4 + 1] = bsrc[e4 + 1];
            lds_bias[e4 + 2] = bsrc[e4 + 2];
            lds_bias[e4 + 3] = bsrc[e4 + 3];
#endif
        }

        const _Float16* Qbase = Qh + (((size_t)(b * NH + h)) * NSEQ + i0) * HD;
        const _Float16* Kbase = Kh + (((size_t)(b * NH + h)) * NSEQ) * HD;
        const _Float16* arow  = Qbase + (size_t)lh * HD;

        // ---- phase 1: logits strip via WMMA (each wave: 4 j-tiles) ----
        #pragma unroll
        for (int q = 0; q < 4; ++q) {
            int jt = wid + q * 8;
            int j0 = jt * 16;
            const _Float16* brow = Kbase + ((size_t)(j0 + lh)) * HD;
            v8f c = {};
            #pragma unroll
            for (int kc = 0; kc < 2; ++kc) {   // HD = 64 = 2 x K32
                v16h a = load_frag(arow, kc, g);
                v16h bb = load_frag(brow, kc, g);
                c = __builtin_amdgcn_wmma_f32_16x16x32_f16(false, a, false, bb, (short)0, c,
                                                           false, false);
            }
            #pragma unroll
            for (int v = 0; v < 8; ++v) {
                int row = v + 8 * g;
                lds_logits[row * NSEQ + j0 + lh] = c[v] * 0.125f;   // 1/sqrt(64)
            }
        }
#if USE_ASYNC_LDS
#if __has_builtin(__builtin_amdgcn_s_wait_asynccnt)
        __builtin_amdgcn_s_wait_asynccnt(0);
#else
        asm volatile("s_wait_asynccnt 0x0" ::: "memory");
#endif
#endif
        __syncthreads();

        // ---- phase 2: softmax over j + accumulate s += w * vp ----
        const float* vph = vp + ((size_t)(b * NH + h)) * NSEQ;
        float vals[32];
        float mx = -3.0e38f;
        #pragma unroll
        for (int k = 0; k < 32; ++k) {
            int j = cb + (k << 4);
            float l = lds_logits[rr * NSEQ + j] + lds_bias[rr * NSEQ + j];
            if (!((mbits >> k) & 1u)) l = -1.0e9f;
            vals[k] = l;
            mx = fmaxf(mx, l);
        }
        #pragma unroll
        for (int off = 1; off < 16; off <<= 1) mx = fmaxf(mx, __shfl_xor(mx, off, 32));
        float se = 0.f;
        #pragma unroll
        for (int k = 0; k < 32; ++k) {
            float e = __expf(vals[k] - mx);
            vals[k] = e;
            se += e;
        }
        #pragma unroll
        for (int off = 1; off < 16; off <<= 1) se += __shfl_xor(se, off, 32);
        float inv = 1.0f / se;
        #pragma unroll
        for (int k = 0; k < 32; ++k) {
            int j = cb + (k << 4);
            lds_s[rr * NSEQ + j] += vals[k] * inv * vph[j];
        }
        __syncthreads();
    }

    // ---- phase 3: force[b,i,c] = sum_j r[b,c,i,j] * s[i,j] + bo (r: stream NT) ----
    int grp = tid >> 4, idx = tid & 15;
    for (int pair = grp; pair < 48; pair += 16) {
        int pr = pair / 3, cc = pair % 3;
        const float* rrow = rin + ((((size_t)(b * 3 + cc)) * NSEQ) + (i0 + pr)) * NSEQ;
        float acc = 0.f;
        #pragma unroll
        for (int k = 0; k < 32; ++k) {
            int j = idx + (k << 4);
            acc += __builtin_nontemporal_load(&rrow[j]) * lds_s[pr * NSEQ + j];
        }
        #pragma unroll
        for (int off = 1; off < 16; off <<= 1) acc += __shfl_xor(acc, off, 32);
        if (idx == 0) out[(((size_t)b * NSEQ + i0 + pr)) * 3 + cc] = acc + bo[0];
    }
}

// ---------------------------------------------------------------------------
extern "C" void kernel_launch(void* const* d_in, const int* in_sizes, int n_in,
                              void* d_out, int out_size, void* d_ws, size_t ws_size,
                              hipStream_t stream) {
    (void)in_sizes; (void)n_in; (void)out_size; (void)ws_size;

    const float* x    = (const float*)d_in[0];
    const float* rin  = (const float*)d_in[1];
    const float* bias = (const float*)d_in[2];
    const int*   mask = (const int*)d_in[3];
    const float* wq   = (const float*)d_in[4];
    const float* bq   = (const float*)d_in[5];
    const float* wk   = (const float*)d_in[6];
    const float* bk   = (const float*)d_in[7];
    const float* wv   = (const float*)d_in[8];
    const float* bv   = (const float*)d_in[9];
    const float* wo   = (const float*)d_in[10];
    const float* bo   = (const float*)d_in[11];
    float* out = (float*)d_out;

    char* ws = (char*)d_ws;
    _Float16* Qh   = (_Float16*)(ws);                           // 4 MB
    _Float16* Kh   = (_Float16*)(ws + ((size_t)4  << 20));      // 4 MB
    _Float16* xh   = (_Float16*)(ws + ((size_t)8  << 20));      // 4 MB
    _Float16* wqT  = (_Float16*)(ws + ((size_t)12 << 20));      // 2 MB
    _Float16* wkT  = (_Float16*)(ws + ((size_t)14 << 20));      // 2 MB
    _Float16* wvph = (_Float16*)(ws + ((size_t)16 << 20));      // 32 KB
    float*    bvp  = (float*)   (ws + ((size_t)16 << 20) + 32768);
    float*    vp   = (float*)   (ws + ((size_t)16 << 20) + 32768 + 256); // 128 KB

    const int NX = BS * NSEQ * DMODEL;                          // 2M elements
    cvt_x_kernel<<<NX / 256, 256, 0, stream>>>(x, xh, NX);
    cvt_wT_kernel<<<(DMODEL * DMODEL) / 256, 256, 0, stream>>>(wq, wqT);
    cvt_wT_kernel<<<(DMODEL * DMODEL) / 256, 256, 0, stream>>>(wk, wkT);
    fold_wv_kernel<<<(NH * DMODEL) / 256, 256, 0, stream>>>(wv, wo, bv, wvph, bvp);

    // Q/K projections: 2048 wave-tiles (32x32 each), 8 waves/block -> 256 blocks
    proj_kernel<<<256, 256, 0, stream>>>(xh, wqT, bq, Qh);
    proj_kernel<<<256, 256, 0, stream>>>(xh, wkT, bk, Kh);

    // vp GEMM: 128 wave-tiles -> 16 blocks
    vp_wmma_kernel<<<16, 256, 0, stream>>>(xh, wvph, bvp, vp);

    attn_kernel<<<BS * (NSEQ / 16), 256, 0, stream>>>(Qh, Kh, vp, bias, mask, rin, bo, out);
}
